// MultiheadAttention_17626545783739
// MI455X (gfx1250) — matrix operand order repaired and checked by reading, not compiled, not run on hardware
//
#include <hip/hip_runtime.h>
#include <stdint.h>

// ---------------------------------------------------------------------------
// MHA forward for MI455X (gfx1250), wave32 + WMMA bf16 + Tensor Data Mover.
// B=4, S=2048, D=1024, H=16, DV=64.
// ---------------------------------------------------------------------------

typedef __bf16 bf16;
typedef __attribute__((ext_vector_type(8)))  bf16  v8bf;
typedef __attribute__((ext_vector_type(16))) bf16  v16bf;
typedef __attribute__((ext_vector_type(8)))  float v8f;
typedef uint32_t u32x4 __attribute__((ext_vector_type(4)));
typedef int      i32x4 __attribute__((ext_vector_type(4)));
typedef int      i32x8 __attribute__((ext_vector_type(8)));

constexpr int BB = 4;
constexpr int SS = 2048;
constexpr int DD = 1024;
constexpr int HH = 16;
constexpr int DV = 64;

#if defined(__has_builtin)
#if __has_builtin(__builtin_amdgcn_tensor_load_to_lds) && \
    __has_builtin(__builtin_amdgcn_s_wait_tensorcnt)
#define USE_TDM 1
#endif
#endif

// ---------------------------------------------------------------------------
// f32 -> bf16 conversion (grid-stride)
// ---------------------------------------------------------------------------
__global__ void k_cvt_bf16(const float* __restrict__ src, bf16* __restrict__ dst, int n) {
  for (int i = blockIdx.x * blockDim.x + threadIdx.x; i < n; i += gridDim.x * blockDim.x)
    dst[i] = (bf16)src[i];
}

// ---------------------------------------------------------------------------
// TDM: async-load a 2D tile (tile_d1 rows x tile_d0 bf16 elems) from global
// (row stride = stride_d0 elems) into LDS, inserting 8 DWORDs of pad after
// every 16 DWORDs (64B row -> 96B LDS stride = LDT elems).
// D# bit layout per CDNA5 ISA 8.3/8.4.
// ---------------------------------------------------------------------------
#ifdef USE_TDM
__device__ __forceinline__ void tdm_load_tile(void* lds_dst, const bf16* gsrc,
                                              uint32_t tile_d0, uint32_t tile_d1,
                                              uint32_t stride_d0) {
  const uint64_t ga = (uint64_t)(uintptr_t)gsrc;
  u32x4 g0;
  g0[0] = 1u;                                            // count=1, user D#
  g0[1] = (uint32_t)(uintptr_t)lds_dst;                  // lds_addr
  g0[2] = (uint32_t)ga;                                  // global_addr[31:0]
  g0[3] = (uint32_t)((ga >> 32) & 0x01FFFFFFu) | (2u << 30);  // addr[56:32] | type=2

  i32x8 g1;
  // data_size=2B(1) | pad_enable | pad_interval=3 (16 DW) | pad_amount=7 (8 DW)
  g1[0] = (int)((1u << 16) | (1u << 20) | (3u << 22) | (7u << 25));
  g1[1] = (int)((tile_d0 & 0xFFFFu) << 16);              // tensor_dim0 = tile_d0
  g1[2] = (int)((tile_d0 >> 16) | ((tile_d1 & 0xFFFFu) << 16));  // tensor_dim1
  g1[3] = (int)((tile_d1 >> 16) | (tile_d0 << 16));      // tile_dim0
  g1[4] = (int)(tile_d1 & 0xFFFFu);                      // tile_dim1 (tile_dim2=0)
  g1[5] = (int)stride_d0;                                // tensor_dim0_stride[31:0]
  g1[6] = 0;                                             // stride hi / dim1_stride lo
  g1[7] = 0;

  i32x4 z4 = {0, 0, 0, 0};
#if __clang_major__ >= 23
  i32x8 z8 = {0, 0, 0, 0, 0, 0, 0, 0};
  __builtin_amdgcn_tensor_load_to_lds(g0, g1, z4, z4, z8, 0);
#else
  __builtin_amdgcn_tensor_load_to_lds(g0, g1, z4, z4, 0);
#endif
}
#endif

// ---------------------------------------------------------------------------
// NT GEMM:  C[M,N] = A[M,K] * Bt[N,K]^T + bias[N],  M=8192, N=K=1024.
// 256 threads = 8 waves; block tile 128x128; K-slab 32 (one WMMA k-step).
// TDM double-buffered staging: wave0 issues tensor_load_to_lds for the next
// slab while all 8 waves compute the current one.
// MODE 1: f32  row-major [M,N]
// MODE 2: bf16 scatter [B,H,S,DV]
// MODE 3: bf16 scatter [B,H,DV,S]  (V transposed for the P*V NT GEMM)
// ---------------------------------------------------------------------------
constexpr int BM = 128, BN = 128, BK = 32, LDT = 48;  // 96B LDS row stride

template <int MODE>
__global__ __launch_bounds__(256)
void k_gemm_nt(const bf16* __restrict__ A, const bf16* __restrict__ Bt,
               const float* __restrict__ bias,
               bf16* __restrict__ outB, float* __restrict__ outF)
{
  constexpr int M = BB * SS, N = DD, K = DD;
  __shared__ __align__(16) bf16 sA[2][BM * LDT];
  __shared__ __align__(16) bf16 sB[2][BN * LDT];

  const int tid  = threadIdx.x;
  const int lane = tid & 31;
  const int wave = tid >> 5;
  const int wm   = wave & 3;   // 4 waves along M
  const int wn   = wave >> 2;  // 2 waves along N
  const int l16  = lane & 15;
  const int grp  = lane >> 4;
  const int m0   = blockIdx.y * BM;
  const int n0   = blockIdx.x * BN;

  v8f acc[2][4];
#pragma unroll
  for (int mt = 0; mt < 2; ++mt)
#pragma unroll
    for (int nt = 0; nt < 4; ++nt)
#pragma unroll
      for (int v = 0; v < 8; ++v) acc[mt][nt][v] = 0.0f;

#ifdef USE_TDM
  if (wave == 0) {
    tdm_load_tile(&sA[0][0], &A[(size_t)m0 * K], BK, BM, K);
    tdm_load_tile(&sB[0][0], &Bt[(size_t)n0 * K], BK, BN, K);
  }
#endif

  int buf = 0;
  for (int k0 = 0; k0 < K; k0 += BK) {
#ifdef USE_TDM
    const bool more = (k0 + BK) < K;
    if (wave == 0) {
      if (more) {
        tdm_load_tile(&sA[buf ^ 1][0], &A[(size_t)m0 * K + k0 + BK], BK, BM, K);
        tdm_load_tile(&sB[buf ^ 1][0], &Bt[(size_t)n0 * K + k0 + BK], BK, BN, K);
        __builtin_amdgcn_s_wait_tensorcnt(2);   // current slab's 2 ops done
      } else {
        __builtin_amdgcn_s_wait_tensorcnt(0);
      }
    }
    __syncthreads();
#else
    // Fallback: cooperative staging, 512 b128 chunks per tile.
#pragma unroll
    for (int c = 0; c < 2; ++c) {
      int ch = tid + c * 256;
      int r  = ch >> 2, cc = (ch & 3) * 8;
      *(v8bf*)&sA[buf][r * LDT + cc] = *(const v8bf*)&A[(size_t)(m0 + r) * K + k0 + cc];
      *(v8bf*)&sB[buf][r * LDT + cc] = *(const v8bf*)&Bt[(size_t)(n0 + r) * K + k0 + cc];
    }
    __syncthreads();
#endif

    v16bf afr[2], bfr[4];
#pragma unroll
    for (int mt = 0; mt < 2; ++mt) {
      int row = wm * 32 + mt * 16 + l16;
      v8bf lo = *(const v8bf*)&sA[buf][row * LDT + grp * 8];
      v8bf hi = *(const v8bf*)&sA[buf][row * LDT + 16 + grp * 8];
#pragma unroll
      for (int i = 0; i < 8; ++i) { afr[mt][i] = lo[i]; afr[mt][8 + i] = hi[i]; }
    }
#pragma unroll
    for (int nt = 0; nt < 4; ++nt) {
      int col = wn * 64 + nt * 16 + l16;
      v8bf lo = *(const v8bf*)&sB[buf][col * LDT + grp * 8];
      v8bf hi = *(const v8bf*)&sB[buf][col * LDT + 16 + grp * 8];
#pragma unroll
      for (int i = 0; i < 8; ++i) { bfr[nt][i] = lo[i]; bfr[nt][8 + i] = hi[i]; }
    }

#pragma unroll
    for (int mt = 0; mt < 2; ++mt)
#pragma unroll
      for (int nt = 0; nt < 4; ++nt)
        acc[mt][nt] = __builtin_amdgcn_wmma_f32_16x16x32_bf16(
            false, afr[mt], false, bfr[nt], (short)0, acc[mt][nt], false, false);

    __syncthreads();
    buf ^= 1;
  }

  // Straight-line epilogue (MODE is compile-time; bias unconditional).
#pragma unroll
  for (int mt = 0; mt < 2; ++mt)
#pragma unroll
    for (int nt = 0; nt < 4; ++nt) {
      const int n  = n0 + wn * 64 + nt * 16 + l16;
      const float bv = bias[n];
#pragma unroll
      for (int v = 0; v < 8; ++v) {
        const int m = m0 + wm * 32 + mt * 16 + v + 8 * grp;
        const float val = acc[mt][nt][v] + bv;
        if constexpr (MODE == 1) {
          outF[(size_t)m * N + n] = val;
        } else {
          const int b = m >> 11, s = m & (SS - 1);      // SS = 2048
          const int h = n >> 6,  d = n & (DV - 1);      // DV = 64
          size_t idx;
          if constexpr (MODE == 2)
            idx = ((size_t)(b * HH + h) * SS + s) * DV + d;   // [B,H,S,DV]
          else
            idx = ((size_t)(b * HH + h) * DV + d) * SS + s;   // [B,H,DV,S]
          outB[idx] = (bf16)val;
        }
      }
    }
}

// ---------------------------------------------------------------------------
// Flash attention: block = 4 waves, handles (b, h, 64 q-rows).
// ---------------------------------------------------------------------------
__global__ __launch_bounds__(128)
void k_flash(const bf16* __restrict__ Qp, const bf16* __restrict__ Kp,
             const bf16* __restrict__ Vt, const uint8_t* __restrict__ mask,
             bf16* __restrict__ ctx)
{
  __shared__ __align__(16) bf16 sP[4 * 16 * 64];

  const int lane = threadIdx.x & 31;
  const int wave = threadIdx.x >> 5;
  const int l16  = lane & 15;
  const int grp  = lane >> 4;
  const int qt   = blockIdx.x;
  const int h    = blockIdx.y;
  const int b    = blockIdx.z;

  const size_t headOff = (size_t)(b * HH + h) * SS * DV;
  const bf16* Qh = Qp + headOff;   // [S, DV]
  const bf16* Kh = Kp + headOff;   // [S, DV]
  const bf16* Vh = Vt + headOff;   // [DV, S]

  const int qrow = qt * 64 + wave * 16 + l16;
  v16bf qf[2];
#pragma unroll
  for (int ks = 0; ks < 2; ++ks) {
    v8bf lo = *(const v8bf*)&Qh[(size_t)qrow * DV + ks * 32 + grp * 8];
    v8bf hi = *(const v8bf*)&Qh[(size_t)qrow * DV + ks * 32 + 16 + grp * 8];
#pragma unroll
    for (int i = 0; i < 8; ++i) { qf[ks][i] = lo[i]; qf[ks][8 + i] = hi[i]; }
  }

  v8f o[4];
  float rowM[8], rowL[8];
#pragma unroll
  for (int nt = 0; nt < 4; ++nt)
#pragma unroll
    for (int v = 0; v < 8; ++v) o[nt][v] = 0.0f;
#pragma unroll
  for (int v = 0; v < 8; ++v) { rowM[v] = -__builtin_inff(); rowL[v] = 0.0f; }

  bf16* myP = sP + wave * 16 * 64;
  const float scale = 0.125f;   // 1/sqrt(DV)

  for (int j = 0; j < SS / 64; ++j) {
    const int kbase = j * 64;
    if (j + 1 < SS / 64) {
      __builtin_prefetch(&Kh[(size_t)(kbase + 64) * DV + lane * 16], 0, 1);
      __builtin_prefetch(&Vh[(size_t)(lane & 15) * SS + kbase + 64], 0, 1);
    }

    // scores S = Q * K^T
    v8f s[4];
#pragma unroll
    for (int nt = 0; nt < 4; ++nt) {
      v8f a;
#pragma unroll
      for (int v = 0; v < 8; ++v) a[v] = 0.0f;
      int kp = kbase + nt * 16 + l16;
#pragma unroll
      for (int ks = 0; ks < 2; ++ks) {
        v8bf lo = *(const v8bf*)&Kh[(size_t)kp * DV + ks * 32 + grp * 8];
        v8bf hi = *(const v8bf*)&Kh[(size_t)kp * DV + ks * 32 + 16 + grp * 8];
        v16bf kf;
#pragma unroll
        for (int i = 0; i < 8; ++i) { kf[i] = lo[i]; kf[8 + i] = hi[i]; }
        a = __builtin_amdgcn_wmma_f32_16x16x32_bf16(false, qf[ks], false, kf,
                                                    (short)0, a, false, false);
      }
      s[nt] = a;
    }

    // scale + mask + online softmax (width-16 butterflies match C layout rows)
#pragma unroll
    for (int v = 0; v < 8; ++v) {
      const int q = qt * 64 + wave * 16 + v + 8 * grp;
      float cand = -__builtin_inff();
#pragma unroll
      for (int nt = 0; nt < 4; ++nt) {
        float sv = s[nt][v] * scale;
        int kk = kbase + nt * 16 + l16;
        if (mask[(size_t)q * SS + kk]) sv = 1e-9f;   // faithful to reference
        s[nt][v] = sv;
        cand = fmaxf(cand, sv);
      }
#pragma unroll
      for (int off = 1; off < 16; off <<= 1)
        cand = fmaxf(cand, __shfl_xor(cand, off, 16));
      const float mnew = fmaxf(rowM[v], cand);
      const float corr = __expf(rowM[v] - mnew);
      float psum = 0.0f;
#pragma unroll
      for (int nt = 0; nt < 4; ++nt) {
        float p = __expf(s[nt][v] - mnew);
        s[nt][v] = p;
        psum += p;
      }
#pragma unroll
      for (int off = 1; off < 16; off <<= 1)
        psum += __shfl_xor(psum, off, 16);
      rowL[v] = rowL[v] * corr + psum;
      rowM[v] = mnew;
#pragma unroll
      for (int nt = 0; nt < 4; ++nt) o[nt][v] *= corr;
    }

    // P: C-layout -> LDS -> A-layout (in-wave, DS ops stay ordered)
#pragma unroll
    for (int nt = 0; nt < 4; ++nt)
#pragma unroll
      for (int v = 0; v < 8; ++v)
        myP[(v + 8 * grp) * 64 + nt * 16 + l16] = (bf16)s[nt][v];

    v16bf pf[2];
#pragma unroll
    for (int ks = 0; ks < 2; ++ks) {
      v8bf lo = *(const v8bf*)&myP[l16 * 64 + ks * 32 + grp * 8];
      v8bf hi = *(const v8bf*)&myP[l16 * 64 + ks * 32 + 16 + grp * 8];
#pragma unroll
      for (int i = 0; i < 8; ++i) { pf[ks][i] = lo[i]; pf[ks][8 + i] = hi[i]; }
    }

    // O += P * V
#pragma unroll
    for (int nt = 0; nt < 4; ++nt) {
      int d = nt * 16 + l16;
#pragma unroll
      for (int ks = 0; ks < 2; ++ks) {
        v8bf lo = *(const v8bf*)&Vh[(size_t)d * SS + kbase + ks * 32 + grp * 8];
        v8bf hi = *(const v8bf*)&Vh[(size_t)d * SS + kbase + ks * 32 + 16 + grp * 8];
        v16bf vf;
#pragma unroll
        for (int i = 0; i < 8; ++i) { vf[i] = lo[i]; vf[8 + i] = hi[i]; }
        o[nt] = __builtin_amdgcn_wmma_f32_16x16x32_bf16(false, pf[ks], false, vf,
                                                        (short)0, o[nt], false, false);
      }
    }
  }

#pragma unroll
  for (int nt = 0; nt < 4; ++nt)
#pragma unroll
    for (int v = 0; v < 8; ++v) {
      const int q = qt * 64 + wave * 16 + v + 8 * grp;
      const int d = nt * 16 + l16;
      ctx[((size_t)b * SS + q) * DD + h * DV + d] = (bf16)(o[nt][v] / rowL[v]);
    }
}

// ---------------------------------------------------------------------------
// Launch
// ---------------------------------------------------------------------------
extern "C" void kernel_launch(void* const* d_in, const int* in_sizes, int n_in,
                              void* d_out, int out_size, void* d_ws, size_t ws_size,
                              hipStream_t stream) {
  const float*   q    = (const float*)d_in[0];
  const float*   k    = (const float*)d_in[1];
  const float*   v    = (const float*)d_in[2];
  const uint8_t* mask = (const uint8_t*)d_in[3];
  const float*   Wq   = (const float*)d_in[4];
  const float*   bq   = (const float*)d_in[5];
  const float*   Wk   = (const float*)d_in[6];
  const float*   bk   = (const float*)d_in[7];
  const float*   Wv   = (const float*)d_in[8];
  const float*   bv   = (const float*)d_in[9];
  const float*   Wo   = (const float*)d_in[10];
  const float*   bo   = (const float*)d_in[11];
  float* out = (float*)d_out;

  const size_t nBSD = (size_t)BB * SS * DD;
  const size_t nDD  = (size_t)DD * DD;

  bf16* p = (bf16*)d_ws;
  bf16* q16  = p; p += nBSD;
  bf16* k16  = p; p += nBSD;
  bf16* v16  = p; p += nBSD;
  bf16* Wq16 = p; p += nDD;
  bf16* Wk16 = p; p += nDD;
  bf16* Wv16 = p; p += nDD;
  bf16* Wo16 = p; p += nDD;
  bf16* Qp   = p; p += nBSD;   // [B,H,S,DV]
  bf16* Kp   = p; p += nBSD;   // [B,H,S,DV]
  bf16* Vt   = p; p += nBSD;   // [B,H,DV,S]
  bf16* ctx  = p; p += nBSD;   // [B,S,D]

  k_cvt_bf16<<<2048, 256, 0, stream>>>(q,  q16,  (int)nBSD);
  k_cvt_bf16<<<2048, 256, 0, stream>>>(k,  k16,  (int)nBSD);
  k_cvt_bf16<<<2048, 256, 0, stream>>>(v,  v16,  (int)nBSD);
  k_cvt_bf16<<<256,  256, 0, stream>>>(Wq, Wq16, (int)nDD);
  k_cvt_bf16<<<256,  256, 0, stream>>>(Wk, Wk16, (int)nDD);
  k_cvt_bf16<<<256,  256, 0, stream>>>(Wv, Wv16, (int)nDD);
  k_cvt_bf16<<<256,  256, 0, stream>>>(Wo, Wo16, (int)nDD);

  const int M = BB * SS;
  dim3 gg(DD / BN, M / BM);   // (8, 64)
  k_gemm_nt<2><<<gg, 256, 0, stream>>>(q16, Wq16, bq, Qp, nullptr);
  k_gemm_nt<2><<<gg, 256, 0, stream>>>(k16, Wk16, bk, Kp, nullptr);
  k_gemm_nt<3><<<gg, 256, 0, stream>>>(v16, Wv16, bv, Vt, nullptr);

  dim3 gf(SS / 64, HH, BB);   // (32, 16, 4)
  k_flash<<<gf, 128, 0, stream>>>(Qp, Kp, Vt, mask, ctx);

  k_gemm_nt<1><<<gg, 256, 0, stream>>>(ctx, Wo16, bo, nullptr, out);
}
